// InvariantPointAttention_9706626089913
// MI455X (gfx1250) — matrix operand order repaired and checked by reading, not compiled, not run on hardware
//
#include <hip/hip_runtime.h>
#include <hip/hip_bf16.h>

typedef __attribute__((ext_vector_type(16))) _Float16 v16h;
typedef __attribute__((ext_vector_type(8)))  _Float16 v8h;
typedef __attribute__((ext_vector_type(8)))  float    v8f;
typedef __attribute__((ext_vector_type(4)))  int      v4i;

namespace {
constexpr int N   = 384;
constexpr int DS  = 384;
constexpr int DP  = 128;
constexpr int DH  = 16;
constexpr int H   = 12;
constexpr int PV  = 8;
constexpr int NPROJ = 1152;   // 192 (q) + 384 (kv) + 144 (q_pts) + 432 (kv_pts)
constexpr int COUT  = 2112;   // 192 + 96*4 + 1536
constexpr int NN  = N * N;    // 147456
constexpr int KCH = 352;      // K-chunk for k_out LDS staging (11 k-steps)
}

// ---------------- gfx1250 async global->LDS copy (ASYNCcnt-tracked DMA) -------
#if defined(__has_builtin)
#if __has_builtin(__builtin_amdgcn_global_load_async_to_lds_b128)
#define USE_ASYNC_LDS 1
#endif
#if __has_builtin(__builtin_amdgcn_s_wait_asynccnt)
#define HAVE_WAIT_ASYNC 1
#endif
#endif

__device__ __forceinline__ void async_copy_b128(const void* g, void* l) {
#ifdef USE_ASYNC_LDS
  // per-lane: 16B global -> 16B LDS, tracked by ASYNCcnt (no VGPR round trip)
  __builtin_amdgcn_global_load_async_to_lds_b128(
      (__attribute__((address_space(1))) v4i*)(void*)(size_t)(const char*)g,
      (__attribute__((address_space(3))) v4i*)l, 0, 0);
#else
  *(int4*)l = *(const int4*)g;
#endif
}
__device__ __forceinline__ void async_wait_0() {
#ifdef USE_ASYNC_LDS
#ifdef HAVE_WAIT_ASYNC
  __builtin_amdgcn_s_wait_asynccnt(0);
#else
  asm volatile("s_wait_asynccnt 0x0" ::: "memory");
#endif
#endif
}

// ---------------- WMMA helpers (CDNA5 16x16x32 f16, f32 accum) ----------------
// A operand: row-major source, per-lane: m = lane&15, K in {8*kh..} / {16+8*kh..}
__device__ __forceinline__ v16h load_a16(const _Float16* base, int ld) {
  const int l  = threadIdx.x & 31;
  const int m  = l & 15, kh = l >> 4;
  const _Float16* p = base + (size_t)m * ld + 8 * kh;
  v8h lo = *(const v8h*)p;          // K = 8*kh .. 8*kh+7
  v8h hi = *(const v8h*)(p + 16);   // K = 16+8*kh .. 16+8*kh+7
  v16h r;
#pragma unroll
  for (int i = 0; i < 8; ++i) { r[i] = lo[i]; r[i + 8] = hi[i]; }
  return r;
}
// B operand: column-major source (column n has contiguous K), per-lane: n = lane&15
__device__ __forceinline__ v16h load_b16(const _Float16* base, int ld) {
  const int l  = threadIdx.x & 31;
  const int n  = l & 15, kh = l >> 4;
  const _Float16* p = base + (size_t)n * ld + 8 * kh;
  v8h lo = *(const v8h*)p;          // K = 8*kh .. 8*kh+7
  v8h hi = *(const v8h*)(p + 16);   // K = 16+8*kh .. 16+8*kh+7
  v16h r;
#pragma unroll
  for (int i = 0; i < 8; ++i) { r[i] = lo[i]; r[i + 8] = hi[i]; }
  return r;
}
__device__ __forceinline__ v8f wmma_f16(v16h a, v16h b, v8f c) {
  return __builtin_amdgcn_wmma_f32_16x16x32_f16(false, a, false, b, (short)0, c,
                                                false, false);
}

// ---------------- conversion kernels ----------------
__global__ void k_cvt_s(const float* __restrict__ s, _Float16* __restrict__ sh) {
  int i = blockIdx.x * 256 + threadIdx.x;
  if (i < N * DS) sh[i] = (_Float16)s[i];
}

__global__ void k_cvt_z(const float* __restrict__ z, _Float16* __restrict__ zh) {
  int i = blockIdx.x * 256 + threadIdx.x;
  if (i < NN * DP) zh[i] = (_Float16)z[i];
}

// z (q,k,c) f32 -> z_col (q,c,k) f16, 32x32 LDS-tiled transpose
__global__ void k_tr_z(const float* __restrict__ z, _Float16* __restrict__ zc) {
  __shared__ float t[32][33];
  int b = blockIdx.x;
  int q = b / 48, rr = b % 48;
  int kt = rr / 4, ct = rr % 4;
#pragma unroll
  for (int i = 0; i < 4; ++i) {
    int idx = threadIdx.x + i * 256;
    int kk = idx >> 5, cc = idx & 31;
    t[kk][cc] = z[(size_t)q * (N * DP) + (size_t)(kt * 32 + kk) * DP + ct * 32 + cc];
  }
  __syncthreads();
#pragma unroll
  for (int i = 0; i < 4; ++i) {
    int idx = threadIdx.x + i * 256;
    int cc = idx >> 5, kk = idx & 31;
    zc[(size_t)q * (N * DP) + (size_t)(ct * 32 + cc) * N + kt * 32 + kk] =
        (_Float16)t[kk][cc];
  }
}

// concatenated projection weights, column-major (n, k contiguous) + bias concat
__global__ void k_cvt_wcat(const float* __restrict__ Wq,  const float* __restrict__ bq,
                           const float* __restrict__ Wkv, const float* __restrict__ bkv,
                           const float* __restrict__ Wqp, const float* __restrict__ bqp,
                           const float* __restrict__ Wkvp,const float* __restrict__ bkvp,
                           _Float16* __restrict__ wc, float* __restrict__ bc) {
  int idx = blockIdx.x * 256 + threadIdx.x;
  if (idx >= NPROJ * DS) return;
  int nc = idx / DS, k = idx % DS;
  float v, b;
  if (nc < 192)      { v = Wq  [(size_t)k * 192 + nc];        b = bq  [nc];       }
  else if (nc < 576) { v = Wkv [(size_t)k * 384 + nc - 192];  b = bkv [nc - 192]; }
  else if (nc < 720) { v = Wqp [(size_t)k * 144 + nc - 576];  b = bqp [nc - 576]; }
  else               { v = Wkvp[(size_t)k * 432 + nc - 720];  b = bkvp[nc - 720]; }
  wc[(size_t)nc * DS + k] = (_Float16)v;
  if (k == 0) bc[nc] = b;
}

// Wb^T (16x128) with sqrt(1/3) folded; rows >= 12 zero
__global__ void k_cvt_wb(const float* __restrict__ Wb, _Float16* __restrict__ wbt) {
  int idx = blockIdx.x * 256 + threadIdx.x;
  if (idx >= 16 * DP) return;
  int h = idx / DP, c = idx % DP;
  wbt[idx] = (h < H) ? (_Float16)(0.5773502691896258f * Wb[(size_t)c * H + h])
                     : (_Float16)0.f;
}

// Wout (COUT, DS) row-major -> column-major (n, k contiguous)
__global__ void k_cvt_wout(const float* __restrict__ Wout, _Float16* __restrict__ wo) {
  int idx = blockIdx.x * 256 + threadIdx.x;
  if (idx >= DS * COUT) return;
  int n = idx / COUT, k = idx % COUT;
  wo[idx] = (_Float16)Wout[(size_t)k * DS + n];
}

// ---------------- projection GEMM: proj = s_h @ Wcat + bias ----------------
// One block = 8 waves sharing one 16-row A block; A staged once via async DMA.
__global__ void __launch_bounds__(256)
k_proj(const _Float16* __restrict__ sh, const _Float16* __restrict__ wc,
       const float* __restrict__ bc, float* __restrict__ proj) {
  __shared__ _Float16 As[16 * DS];  // 12 KB: rows m0..m0+15, all K
  const int wave = threadIdx.x >> 5;
  const int m0 = (int)(blockIdx.x / 9) * 16;
  const int n0 = ((int)(blockIdx.x % 9) * 8 + wave) * 16;

  {  // cooperative async stage: 12288 contiguous bytes, 16B per lane per issue
    const char* g = (const char*)(sh + (size_t)m0 * DS);
    char* l = (char*)As;
#pragma unroll
    for (int ofs = 0; ofs < 16 * DS * 2; ofs += 256 * 16)
      async_copy_b128(g + ofs + threadIdx.x * 16, l + ofs + threadIdx.x * 16);
    async_wait_0();
  }
  __syncthreads();

  v8f c = {};
  for (int k0 = 0; k0 < DS; k0 += 32) {
    v16h a = load_a16(As + k0, DS);
    v16h b = load_b16(wc + (size_t)n0 * DS + k0, DS);
    c = wmma_f16(a, b, c);
  }
  int l = threadIdx.x & 31, kh = l >> 4, nl = l & 15;
#pragma unroll
  for (int r = 0; r < 8; ++r) {
    int m = m0 + r + 8 * kh, n = n0 + nl;
    proj[(size_t)m * NPROJ + n] = c[r] + bc[n];
  }
}

// ---------------- feature build: frames, augmented q/k features ----------------
__global__ void k_features(const float* __restrict__ proj, const float* __restrict__ rot,
                           const float* __restrict__ trans, const float* __restrict__ head_w,
                           _Float16* __restrict__ qfeat, _Float16* __restrict__ kfeat,
                           float* __restrict__ rowq, float* __restrict__ rowk,
                           _Float16* __restrict__ vT, _Float16* __restrict__ vp) {
  int idx = blockIdx.x * 256 + threadIdx.x;
  if (idx >= N * H) return;
  int n = idx / H, h = idx % H;
  const float* pr = proj + (size_t)n * NPROJ;
  float R[9], T[3];
#pragma unroll
  for (int i = 0; i < 9; ++i) R[i] = rot[n * 9 + i];
#pragma unroll
  for (int i = 0; i < 3; ++i) T[i] = trans[n * 3 + i];
  // hw = softplus(w) * sqrt(1/(3*(PQK*9/2))) = softplus(w)*sqrt(1/54)
  float hw = log1pf(__expf(head_w[h])) * 0.13608276348795434f;
  const float s1 = 0.14433756729740643f;  // sqrt(1/(3*DH))

  size_t qb = ((size_t)h * N + n) * 32;
#pragma unroll
  for (int d = 0; d < 16; ++d) {
    float qv = pr[h * 16 + d];
    float kv = pr[192 + h * 32 + d];
    float vv = pr[192 + h * 32 + 16 + d];
    qfeat[qb + d] = (_Float16)(qv * s1);
    kfeat[qb + d] = (_Float16)kv;
    vT[((size_t)h * 16 + d) * N + n] = (_Float16)vv;
  }
  float sq = 0.f;
  for (int p = 0; p < 4; ++p) {
    float l0 = pr[576 + 0 * 48 + h * 4 + p];
    float l1 = pr[576 + 1 * 48 + h * 4 + p];
    float l2 = pr[576 + 2 * 48 + h * 4 + p];
    for (int i = 0; i < 3; ++i) {
      float pt = R[i * 3 + 0] * l0 + R[i * 3 + 1] * l1 + R[i * 3 + 2] * l2 + T[i];
      qfeat[qb + 16 + p * 3 + i] = (_Float16)(pt * hw);
      sq += pt * pt;
    }
  }
  rowq[h * N + n] = -0.5f * hw * sq;
  float sk = 0.f;
  for (int p = 0; p < 12; ++p) {
    float l0 = pr[720 + 0 * 144 + h * 12 + p];
    float l1 = pr[720 + 1 * 144 + h * 12 + p];
    float l2 = pr[720 + 2 * 144 + h * 12 + p];
    float pt3[3];
    for (int i = 0; i < 3; ++i)
      pt3[i] = R[i * 3 + 0] * l0 + R[i * 3 + 1] * l1 + R[i * 3 + 2] * l2 + T[i];
    if (p < 4) {
      for (int i = 0; i < 3; ++i) {
        kfeat[qb + 16 + p * 3 + i] = (_Float16)pt3[i];
        sk += pt3[i] * pt3[i];
      }
    } else {
      int pv = p - 4;
      for (int i = 0; i < 3; ++i)
        vp[((size_t)h * 32 + i * 8 + pv) * N + n] = (_Float16)pt3[i];
    }
  }
  rowk[h * N + n] = -0.5f * hw * sk;
  for (int f = 28; f < 32; ++f) { qfeat[qb + f] = (_Float16)0.f; kfeat[qb + f] = (_Float16)0.f; }
  for (int cc = 24; cc < 32; ++cc) vp[((size_t)h * 32 + cc) * N + n] = (_Float16)0.f;
}

// ---------------- bias GEMM: bias_t(h,q,k) = sqrt(1/3)*(z@Wb + bb) via Wb^T @ z^T ----
__global__ void k_biasgemm(const _Float16* __restrict__ wbt, const _Float16* __restrict__ zh,
                           const float* __restrict__ bb, float* __restrict__ bias_t) {
  int tile = blockIdx.x * 8 + (threadIdx.x >> 5);
  int n0 = tile * 16;  // flat (q*N+k) column tile
  __builtin_prefetch(zh + (size_t)(n0 + 128) * DP, 0, 0);  // next blocks' stream
  v8f c = {};
  for (int k0 = 0; k0 < DP; k0 += 32) {
    v16h a = load_a16(wbt + k0, DP);                      // A = Wb^T (16x128)
    v16h b = load_b16(zh + (size_t)n0 * DP + k0, DP);     // B cols = z rows (c contig)
    c = wmma_f16(a, b, c);
  }
  int l = threadIdx.x & 31, kh = l >> 4, nl = l & 15;
#pragma unroll
  for (int r = 0; r < 8; ++r) {
    int h = r + 8 * kh;
    if (h < H)
      bias_t[(size_t)h * NN + n0 + nl] = c[r] + 0.5773502691896258f * bb[h];
  }
}

// ---------------- attention core: one wave per (h, 16-q tile) ----------------
__global__ void __launch_bounds__(32)
k_attn(const _Float16* __restrict__ qfeat, const _Float16* __restrict__ kfeat,
       const float* __restrict__ rowq, const float* __restrict__ rowk,
       const float* __restrict__ bias_t, const float* __restrict__ mask,
       const _Float16* __restrict__ vT, const _Float16* __restrict__ vp,
       _Float16* __restrict__ cat_h, _Float16* __restrict__ attn_g,
       float* __restrict__ o_raw) {
  __shared__ float    lg[16 * 384];
  __shared__ _Float16 at[16 * 384];
  const int h  = blockIdx.x / 24;
  const int q0 = (blockIdx.x % 24) * 16;
  const int l = threadIdx.x & 31, kh = l >> 4, nl = l & 15;

  // logits: single WMMA per k-tile fuses QK^T and point cross term (K=32)
  v16h aq = load_a16(qfeat + ((size_t)h * N + q0) * 32, 32);
  for (int kt = 0; kt < 24; ++kt) {
    v16h bk = load_b16(kfeat + ((size_t)h * N + kt * 16) * 32, 32);
    v8f c = {};
    c = wmma_f16(aq, bk, c);
#pragma unroll
    for (int r = 0; r < 8; ++r) {
      int lr = r + 8 * kh;
      int m = q0 + lr, n = kt * 16 + nl;
      lg[lr * 384 + n] = c[r] + rowq[h * N + m] + rowk[h * N + n] +
                         bias_t[(size_t)h * NN + (size_t)m * N + n] +
                         mask[(size_t)m * N + n];
    }
  }
  __syncthreads();
  // softmax: 2 lanes per row, 192 cols each, combine via shfl_xor(16)
  {
    int lr = nl, c0 = kh * 192;
    float mx = -3.0e38f;
    for (int c = 0; c < 192; ++c) mx = fmaxf(mx, lg[lr * 384 + c0 + c]);
    mx = fmaxf(mx, __shfl_xor(mx, 16));
    float sum = 0.f;
    for (int c = 0; c < 192; ++c) sum += __expf(lg[lr * 384 + c0 + c] - mx);
    sum += __shfl_xor(sum, 16);
    float inv = 1.f / sum;
    for (int c = 0; c < 192; ++c) {
      float a = __expf(lg[lr * 384 + c0 + c] - mx) * inv;
      at[lr * 384 + c0 + c] = (_Float16)a;
      attn_g[(size_t)(q0 + lr) * (H * N) + (size_t)h * N + c0 + c] = (_Float16)a;
    }
  }
  __syncthreads();
  // o = attn @ v (B = v^T, K=384)
  {
    v8f c = {};
    for (int kt = 0; kt < 12; ++kt) {
      v16h a = load_a16(at + kt * 32, 384);
      v16h b = load_b16(vT + (size_t)h * DH * N + kt * 32, N);
      c = wmma_f16(a, b, c);
    }
#pragma unroll
    for (int r = 0; r < 8; ++r) {
      int m = q0 + r + 8 * kh;
      cat_h[(size_t)m * COUT + h * DH + nl] = (_Float16)c[r];
    }
  }
  // o_pts raw (24 cols padded to 32): two N-tiles
  for (int nt = 0; nt < 2; ++nt) {
    v8f c = {};
    for (int kt = 0; kt < 12; ++kt) {
      v16h a = load_a16(at + kt * 32, 384);
      v16h b = load_b16(vp + ((size_t)h * 32 + nt * 16) * N + kt * 32, N);
      c = wmma_f16(a, b, c);
    }
#pragma unroll
    for (int r = 0; r < 8; ++r) {
      int m = q0 + r + 8 * kh;
      o_raw[(size_t)m * 384 + h * 32 + nt * 16 + nl] = c[r];
    }
  }
}

// ---------------- o_pts inverse frame + norm ----------------
__global__ void k_opts_post(const float* __restrict__ raw, const float* __restrict__ rot,
                            const float* __restrict__ trans, _Float16* __restrict__ cat_h) {
  int idx = blockIdx.x * 256 + threadIdx.x;
  if (idx >= N * H * PV) return;
  int q = idx / (H * PV);
  int r = idx % (H * PV);
  int h = r / PV, p = r % PV;
  float x[3];
  for (int i = 0; i < 3; ++i)
    x[i] = raw[(size_t)q * 384 + h * 32 + i * 8 + p] - trans[q * 3 + i];
  float ss = 1e-8f;
  float o[3];
  for (int i = 0; i < 3; ++i) {
    o[i] = rot[q * 9 + 0 * 3 + i] * x[0] + rot[q * 9 + 1 * 3 + i] * x[1] +
           rot[q * 9 + 2 * 3 + i] * x[2];
    ss += o[i] * o[i];
  }
  for (int i = 0; i < 3; ++i)
    cat_h[(size_t)q * COUT + 192 + i * 96 + h * 8 + p] = (_Float16)o[i];
  cat_h[(size_t)q * COUT + 480 + h * 8 + p] = (_Float16)sqrtf(ss);
}

// ---------------- o_pair: per-q GEMM attn(12x384) @ z_q(384x128) ----------------
__global__ void k_opair(const _Float16* __restrict__ attn_g, const _Float16* __restrict__ z_col,
                        _Float16* __restrict__ cat_h) {
  int q = blockIdx.x * 8 + (threadIdx.x >> 5);
  int l = threadIdx.x & 31, kh = l >> 4, nl = l & 15;
  v8f zero = {};
  v8f acc[8];
#pragma unroll
  for (int i = 0; i < 8; ++i) acc[i] = zero;
  for (int kt = 0; kt < 12; ++kt) {
    __builtin_prefetch(z_col + (size_t)q * (DP * N) + (size_t)(kt + 1) * 32, 0, 0);
    v16h a = load_a16(attn_g + (size_t)q * (H * N) + kt * 32, N);
#pragma unroll
    for (int nt = 0; nt < 8; ++nt) {
      v16h b = load_b16(z_col + (size_t)q * (DP * N) + (size_t)nt * 16 * N + kt * 32, N);
      acc[nt] = wmma_f16(a, b, acc[nt]);
    }
  }
#pragma unroll
  for (int nt = 0; nt < 8; ++nt)
#pragma unroll
    for (int r = 0; r < 8; ++r) {
      int h = r + 8 * kh;
      if (h < H)
        cat_h[(size_t)q * COUT + 576 + h * DP + nt * 16 + nl] = (_Float16)acc[nt][r];
    }
}

// ---------------- final GEMM: out = cat_h @ Wout + bout ----------------
// 8 waves share the 16-row A block; A staged in 352-col chunks via async DMA.
__global__ void __launch_bounds__(256)
k_out(const _Float16* __restrict__ cat_h, const _Float16* __restrict__ wo,
      const float* __restrict__ bout, float* __restrict__ out) {
  __shared__ _Float16 As[16 * KCH];  // 11 KB chunk
  const int wave = threadIdx.x >> 5;
  const int m0 = (int)(blockIdx.x / 3) * 16;
  const int n0 = ((int)(blockIdx.x % 3) * 8 + wave) * 16;

  v8f c = {};
  for (int kc = 0; kc < COUT; kc += KCH) {
    // stage A chunk: 16 rows x 352 cols f16 (44 x 16B per row)
    for (int idx = threadIdx.x; idx < 16 * (KCH / 8); idx += 256) {
      int row = idx / (KCH / 8), seg = idx % (KCH / 8);
      async_copy_b128((const char*)(cat_h + (size_t)(m0 + row) * COUT + kc) + seg * 16,
                      (char*)(As + row * KCH) + seg * 16);
    }
    async_wait_0();
    __syncthreads();
    for (int kk = 0; kk < KCH; kk += 32) {
      v16h a = load_a16(As + kk, KCH);
      v16h b = load_b16(wo + (size_t)n0 * COUT + kc + kk, COUT);
      c = wmma_f16(a, b, c);
    }
    __syncthreads();  // before restaging
  }
  int l = threadIdx.x & 31, kh = l >> 4, nl = l & 15;
#pragma unroll
  for (int r = 0; r < 8; ++r) {
    int m = m0 + r + 8 * kh, n = n0 + nl;
    out[(size_t)m * DS + n] = c[r] + bout[n];
  }
}

// ---------------- host launcher ----------------
extern "C" void kernel_launch(void* const* d_in, const int* in_sizes, int n_in,
                              void* d_out, int out_size, void* d_ws, size_t ws_size,
                              hipStream_t stream) {
  (void)in_sizes; (void)n_in; (void)out_size; (void)ws_size;
  const float* s     = (const float*)d_in[0];
  const float* z     = (const float*)d_in[1];
  const float* rot   = (const float*)d_in[2];
  const float* trans = (const float*)d_in[3];
  const float* mask  = (const float*)d_in[4];
  const float* Wq    = (const float*)d_in[5];
  const float* bq    = (const float*)d_in[6];
  const float* Wkv   = (const float*)d_in[7];
  const float* bkv   = (const float*)d_in[8];
  const float* Wqp   = (const float*)d_in[9];
  const float* bqp   = (const float*)d_in[10];
  const float* Wkvp  = (const float*)d_in[11];
  const float* bkvp  = (const float*)d_in[12];
  const float* Wb    = (const float*)d_in[13];
  const float* bb    = (const float*)d_in[14];
  const float* hwv   = (const float*)d_in[15];
  const float* Wout  = (const float*)d_in[16];
  const float* bout  = (const float*)d_in[17];
  float* out = (float*)d_out;

  char* base = (char*)d_ws;
  size_t off = 0;
  auto carve = [&](size_t bytes) -> void* {
    void* p = base + off;
    off += (bytes + 255) & ~(size_t)255;
    return p;
  };
  _Float16* s_h    = (_Float16*)carve((size_t)N * DS * 2);
  _Float16* wcat   = (_Float16*)carve((size_t)NPROJ * DS * 2);
  float*    bcat   = (float*)   carve((size_t)NPROJ * 4);
  float*    proj   = (float*)   carve((size_t)N * NPROJ * 4);
  _Float16* z_h    = (_Float16*)carve((size_t)NN * DP * 2);
  _Float16* z_col  = (_Float16*)carve((size_t)NN * DP * 2);
  _Float16* wb_t   = (_Float16*)carve((size_t)16 * DP * 2);
  float*    bias_t = (float*)   carve((size_t)H * NN * 4);
  _Float16* qfeat  = (_Float16*)carve((size_t)H * N * 32 * 2);
  _Float16* kfeat  = (_Float16*)carve((size_t)H * N * 32 * 2);
  float*    rowq   = (float*)   carve((size_t)H * N * 4);
  float*    rowk   = (float*)   carve((size_t)H * N * 4);
  _Float16* vT     = (_Float16*)carve((size_t)H * DH * N * 2);
  _Float16* vp     = (_Float16*)carve((size_t)H * 32 * N * 2);
  _Float16* attn_g = (_Float16*)carve(((size_t)N * H * N + 4 * N) * 2); // + pad rows
  float*    o_raw  = (float*)   carve((size_t)N * H * 32 * 4);
  _Float16* cat_h  = (_Float16*)carve((size_t)N * COUT * 2);
  _Float16* wout_c = (_Float16*)carve((size_t)DS * COUT * 2);

  k_cvt_s   <<<(N * DS + 255) / 256, 256, 0, stream>>>(s, s_h);
  k_cvt_wcat<<<(NPROJ * DS + 255) / 256, 256, 0, stream>>>(Wq, bq, Wkv, bkv, Wqp, bqp,
                                                           Wkvp, bkvp, wcat, bcat);
  k_cvt_z   <<<(NN * DP + 255) / 256, 256, 0, stream>>>(z, z_h);
  k_tr_z    <<<N * 48, 256, 0, stream>>>(z, z_col);
  k_cvt_wb  <<<(16 * DP + 255) / 256, 256, 0, stream>>>(Wb, wb_t);
  k_cvt_wout<<<(DS * COUT + 255) / 256, 256, 0, stream>>>(Wout, wout_c);

  k_proj    <<<(24 * 72) / 8, 256, 0, stream>>>(s_h, wcat, bcat, proj);
  k_features<<<(N * H + 255) / 256, 256, 0, stream>>>(proj, rot, trans, hwv, qfeat,
                                                      kfeat, rowq, rowk, vT, vp);
  k_biasgemm<<<(NN / 16) / 8, 256, 0, stream>>>(wb_t, z_h, bb, bias_t);
  k_attn    <<<H * 24, 32, 0, stream>>>(qfeat, kfeat, rowq, rowk, bias_t, mask, vT, vp,
                                        cat_h, attn_g, o_raw);
  k_opts_post<<<(N * H * PV + 255) / 256, 256, 0, stream>>>(o_raw, rot, trans, cat_h);
  k_opair   <<<N / 8, 256, 0, stream>>>(attn_g, z_col, cat_h);
  k_out     <<<(24 * 24) / 8, 256, 0, stream>>>(cat_h, wout_c, bout, out);
}
